// RelativePositionalMultiHeadSelfAttention_59966333387236
// MI455X (gfx1250) — matrix operand order repaired and checked by reading, not compiled, not run on hardware
//
#include <hip/hip_runtime.h>

// ---------------- constants ----------------
#define NB   4
#define CC   64
#define VV   25
#define TT   300
#define MM   2
#define HH   8
#define HD   32
#define EE   256
#define BB   (NB*MM)        // 8
#define TP   320            // padded T (multiple of 32)
#define NT_T 19             // row tiles of 16 covering 304 >= 300
#define BTP  640            // padded bias-table rows (need up to 603)

#if defined(__has_builtin)
#if __has_builtin(__builtin_amdgcn_global_load_async_to_lds_b128) && \
    __has_builtin(__builtin_amdgcn_s_wait_asynccnt)
#define HAVE_ASYNC 1
#endif
#endif
#ifndef HAVE_ASYNC
#define HAVE_ASYNC 0
#endif

typedef __attribute__((ext_vector_type(16))) __bf16 bfrag;
typedef __attribute__((ext_vector_type(8)))  float  v8f;

#if HAVE_ASYNC
typedef int async_v4i __attribute__((vector_size(16)));
typedef __attribute__((address_space(1))) async_v4i* g_v4i_ptr;
typedef __attribute__((address_space(3))) async_v4i* l_v4i_ptr;
#endif

// float -> bf16: native convert (RNE) -> single v_cvt instead of bit games
__device__ inline __bf16 f2bf(float f) { return (__bf16)f; }

// A fragment (16x32 bf16): lane L: row = L%16, g = L/16
// elements 0..7  <-> K = kbase + 8g + 0..7
// elements 8..15 <-> K = kbase + 16 + 8g + 0..7
__device__ inline bfrag load_a_frag(const __bf16* row, int kbase, int g) {
  bfrag a;
  const __bf16* p0 = row + kbase + 8*g;
  const __bf16* p1 = row + kbase + 16 + 8*g;
#pragma unroll
  for (int i = 0; i < 8; ++i) { a[i] = p0[i]; a[8+i] = p1[i]; }
  return a;
}

// B fragment (32x16 bf16): lane L: col = L%16, g = L/16
// elements 0..7 <-> K = kbase + 8g + 0..7, 8..15 <-> K = kbase + 16 + 8g + 0..7  (col stored contiguous in K)
__device__ inline bfrag load_b_frag(const __bf16* col, int kbase, int g) {
  bfrag b;
  const __bf16* p0 = col + kbase + 8*g;
  const __bf16* p1 = col + kbase + 16 + 8*g;
#pragma unroll
  for (int i = 0; i < 8; ++i) { b[i] = p0[i]; b[8+i] = p1[i]; }
  return b;
}

__device__ inline v8f wmma_bf16(bfrag a, bfrag b, v8f c) {
  return __builtin_amdgcn_wmma_f32_16x16x32_bf16(false, a, false, b, (short)0, c,
                                                 false, false);
}

// ---------------- pack kernels ----------------
// xb[bv][t][c] = bf16( x[n][c][v][t][mm] ), zero for t>=300
__global__ void pack_x(const float* __restrict__ x, __bf16* __restrict__ xb) {
  int idx = blockIdx.x * 256 + threadIdx.x;
  const int total = BB * VV * TP * CC;
  if (idx >= total) return;
  int c  = idx & 63;
  int t  = (idx >> 6) % TP;
  int bv = idx / (TP * CC);
  int v  = bv % VV, b = bv / VV;
  int n  = b >> 1, mm = b & 1;
  float val = 0.f;
  if (t < TT)
    val = x[(((((size_t)n*CC + c)*VV + v)*TT + t)*MM) + mm];
  xb[idx] = f2bf(val);
}

// wqkvT[j][c] = bf16(w_qkv[c][j])   (768 x 64)
__global__ void pack_wqkv(const float* __restrict__ w, __bf16* __restrict__ wt) {
  int idx = blockIdx.x * 256 + threadIdx.x;
  if (idx >= 768 * 64) return;
  int c = idx & 63, j = idx >> 6;
  wt[idx] = f2bf(w[(size_t)c * 768 + j]);
}

// btb[j][hd] = bf16(bias_table[j][hd]) for j<599 else 0  (640 x 32)
__global__ void pack_bt(const float* __restrict__ bt, __bf16* __restrict__ btb) {
  int idx = blockIdx.x * 256 + threadIdx.x;
  if (idx >= BTP * HD) return;
  int hd = idx & 31, j = idx >> 5;
  btb[idx] = f2bf((j < 2*TT - 1) ? bt[(size_t)j * HD + hd] : 0.f);
}

// wmT[c][e] = bf16(w_merge[e][c])   (64 x 256)
__global__ void pack_wm(const float* __restrict__ w, __bf16* __restrict__ wt) {
  int idx = blockIdx.x * 256 + threadIdx.x;
  if (idx >= CC * EE) return;
  int e = idx & 255, c = idx >> 8;
  wt[idx] = f2bf(w[(size_t)e * CC + c]);
}

// ---------------- stage 1: QKV projection ----------------
// [B*V*TP x 64] * [64 x 768] + bias -> q/k: [bvh][TP][HD], val: [bvh][HD][TP]
// Each wave computes TWO adjacent 16x16 column tiles (interleaved WMMA accs
// hide the WMMA->WMMA RAW hazard; A fragment shared).
__global__ __launch_bounds__(256)
void qkv_gemm(const __bf16* __restrict__ xb, const __bf16* __restrict__ wt,
              const float* __restrict__ bias,
              __bf16* __restrict__ qb, __bf16* __restrict__ kb,
              __bf16* __restrict__ vt) {
  const int ROWT = BB * VV * NT_T;            // 3800
  int wid  = threadIdx.x >> 5;
  int lane = threadIdx.x & 31;
  int tile = blockIdx.x * 8 + wid;
  if (tile >= ROWT * 24) return;              // wave-uniform
  int jp = tile % 24, rt = tile / 24;
  int tt = rt % NT_T, bv = rt / NT_T;
  int t0 = tt * 16;
  int g = lane >> 4, ln = lane & 15;
  const __bf16* arow = xb + ((size_t)bv * TP + t0 + ln) * CC;
  int j0 = jp * 32 + ln;                      // first col tile
  int j1 = j0 + 16;                           // second col tile
  const __bf16* bcol0 = wt + (size_t)j0 * CC;
  const __bf16* bcol1 = wt + (size_t)j1 * CC;
  v8f acc0 = {}, acc1 = {};
#pragma unroll
  for (int ks = 0; ks < CC; ks += 32) {
    bfrag a  = load_a_frag(arow, ks, g);
    bfrag b0 = load_b_frag(bcol0, ks, g);
    bfrag b1 = load_b_frag(bcol1, ks, g);
    acc0 = wmma_bf16(a, b0, acc0);
    acc1 = wmma_bf16(a, b1, acc1);
  }
  float bj0 = bias[j0], bj1 = bias[j1];
  int h = jp & 7;                             // head (same for both tiles)
  size_t bvh = (size_t)bv * HH + h;
  int region = jp >> 3;                       // wave-uniform: 0=q 1=k 2=v
  if (region == 0) {
#pragma unroll
    for (int r = 0; r < 8; ++r) {
      int t = t0 + g * 8 + r;
      qb[(bvh * TP + t) * HD + ln]      = f2bf(acc0[r] + bj0);
      qb[(bvh * TP + t) * HD + 16 + ln] = f2bf(acc1[r] + bj1);
    }
  } else if (region == 1) {
#pragma unroll
    for (int r = 0; r < 8; ++r) {
      int t = t0 + g * 8 + r;
      kb[(bvh * TP + t) * HD + ln]      = f2bf(acc0[r] + bj0);
      kb[(bvh * TP + t) * HD + 16 + ln] = f2bf(acc1[r] + bj1);
    }
  } else {
#pragma unroll
    for (int r = 0; r < 8; ++r) {
      int t = t0 + g * 8 + r;
      vt[(bvh * HD + ln) * TP + t]        = f2bf(acc0[r] + bj0);
      vt[(bvh * HD + 16 + ln) * TP + t]   = f2bf(acc1[r] + bj1);
    }
  }
}

// ---------------- stage 2: attention (one wave per bvh x 16-row tile) ------
__global__ __launch_bounds__(32)
void attn_kernel(const __bf16* __restrict__ qb, const __bf16* __restrict__ kb,
                 const __bf16* __restrict__ vt, const __bf16* __restrict__ btb,
                 __bf16* __restrict__ ao) {
  __shared__ float qbias[16 * 320];           // QB band: j in [l0, l0+320)
  __shared__ float sc[16 * 320];              // scores / probs
#if HAVE_ASYNC
  __shared__ __bf16 kpanel[TP * HD];          // async-staged K panel (20KB)
#endif
  int lane = threadIdx.x;
  int g = lane >> 4, ln = lane & 15;
  int tt  = blockIdx.x % NT_T;
  int bvh = blockIdx.x / NT_T;
  int bv = bvh >> 3, h = bvh & 7;
  int l0 = tt * 16;

#if HAVE_ASYNC
  // Kick off async HBM->LDS copy of the whole K panel; overlaps with the
  // QB (q . bias_table^T) WMMA phase below. 40 x b128 per lane stream.
  {
    const char* ksrc = (const char*)(kb + (size_t)bvh * TP * HD);
    char* kdst = (char*)kpanel;
#pragma unroll
    for (int i = 0; i < (TP * HD * 2) / (32 * 16); ++i) {   // 40
      int off = (i * 32 + lane) * 16;
      __builtin_amdgcn_global_load_async_to_lds_b128(
          (g_v4i_ptr)(ksrc + off),
          (l_v4i_ptr)(kdst + off), 0, 0);
    }
  }
#endif

  const __bf16* qrow = qb + ((size_t)bvh * TP + l0 + ln) * HD;
  bfrag aq = load_a_frag(qrow, 0, g);         // q tile 16x32 (K=HD)

  // QB = q . bias_table^T for the needed diagonal band
  for (int jt = 0; jt < 20; ++jt) {
    int j = l0 + jt * 16 + ln;                // < 608 < BTP
    bfrag b = load_b_frag(btb + (size_t)j * HD, 0, g);
    v8f d = {};
    d = wmma_bf16(aq, b, d);
#pragma unroll
    for (int r = 0; r < 8; ++r)
      qbias[(g * 8 + r) * 320 + jt * 16 + ln] = d[r];
  }
#if HAVE_ASYNC
  __builtin_amdgcn_s_wait_asynccnt(0);
#endif
  __syncthreads();

  // scores = scale * q.k^T + QB[m][m - r + 299]
  const float scale = 0.0625f;                // E^-0.5
  for (int nt = 0; nt < NT_T; ++nt) {
    int rcol = nt * 16 + ln;
#if HAVE_ASYNC
    const __bf16* krow = kpanel + (size_t)rcol * HD;
#else
    const __bf16* krow = kb + ((size_t)bvh * TP + rcol) * HD;
#endif
    bfrag b = load_b_frag(krow, 0, g);
    v8f d = {};
    d = wmma_bf16(aq, b, d);
#pragma unroll
    for (int r = 0; r < 8; ++r) {
      int m = g * 8 + r;
      float s = -1e30f;
      if (rcol < TT) s = d[r] * scale + qbias[m * 320 + (m - rcol + TT - 1)];
      sc[m * 320 + rcol] = s;
    }
  }
  __syncthreads();

  // row softmax: each row is split across the two half-waves; lane (ln,g)
  // owns columns [160g, 160g+160) of row ln; partials merged via shfl_xor 16.
  {
    float* row = sc + ln * 320;
    int c0 = g * 160;
    float mx = -1e30f;
    for (int i = c0; i < c0 + 160; ++i)
      if (i < TT) mx = fmaxf(mx, row[i]);
    mx = fmaxf(mx, __shfl_xor(mx, 16, 32));
    float sum = 0.f;
    for (int i = c0; i < c0 + 160; ++i) {
      float p = 0.f;
      if (i < TT) p = __expf(row[i] - mx);
      row[i] = p;                              // padded cols -> 0 prob
      sum += p;
    }
    sum += __shfl_xor(sum, 16, 32);
    float inv = 1.f / sum;
    for (int i = c0; i < c0 + 160; ++i) row[i] *= inv;
  }
  __syncthreads();

  // out = attn . val : both 16-wide hd tiles in one k-loop (shared A frag,
  // interleaved accumulators hide WMMA RAW hazard)
  {
    v8f acc0 = {}, acc1 = {};
    const __bf16* bcol0 = vt + ((size_t)bvh * HD + ln) * TP;
    const __bf16* bcol1 = vt + ((size_t)bvh * HD + 16 + ln) * TP;
    for (int kt = 0; kt < TP / 32; ++kt) {
      int k0 = kt * 32;
      bfrag a;
      const float* prow = sc + ln * 320 + k0;
#pragma unroll
      for (int i = 0; i < 8; ++i) {
        a[i]     = f2bf(prow[8 * g + i]);
        a[8 + i] = f2bf(prow[16 + 8 * g + i]);
      }
      bfrag b0 = load_b_frag(bcol0, k0, g);
      bfrag b1 = load_b_frag(bcol1, k0, g);
      acc0 = wmma_bf16(a, b0, acc0);
      acc1 = wmma_bf16(a, b1, acc1);
    }
#pragma unroll
    for (int r = 0; r < 8; ++r) {
      int t = l0 + g * 8 + r;                  // < 304 <= TP
      size_t base = ((size_t)bv * TP + t) * EE + h * HD;
      ao[base + ln]      = f2bf(acc0[r]);
      ao[base + 16 + ln] = f2bf(acc1[r]);
    }
  }
}

// ---------------- stage 3: merge GEMM + output transpose ----------------
// Each wave computes TWO adjacent 16x16 column tiles (c pair) of the
// [60000x256]x[256x64] GEMM.
__global__ __launch_bounds__(256)
void merge_gemm(const __bf16* __restrict__ ao, const __bf16* __restrict__ wt,
                const float* __restrict__ bias, float* __restrict__ out) {
  const int ROWT = BB * VV * NT_T;            // 3800
  int wid  = threadIdx.x >> 5;
  int lane = threadIdx.x & 31;
  int tile = blockIdx.x * 8 + wid;
  if (tile >= ROWT * 2) return;               // wave-uniform
  int cp = tile & 1, rt = tile >> 1;
  int tt = rt % NT_T, bv = rt / NT_T;
  int t0 = tt * 16;
  int g = lane >> 4, ln = lane & 15;
  const __bf16* arow = ao + ((size_t)bv * TP + t0 + ln) * EE;
  int c0 = cp * 32 + ln;
  int c1 = c0 + 16;
  const __bf16* bcol0 = wt + (size_t)c0 * EE;
  const __bf16* bcol1 = wt + (size_t)c1 * EE;
  v8f acc0 = {}, acc1 = {};
#pragma unroll
  for (int ks = 0; ks < EE; ks += 32) {
    bfrag a  = load_a_frag(arow, ks, g);
    bfrag b0 = load_b_frag(bcol0, ks, g);
    bfrag b1 = load_b_frag(bcol1, ks, g);
    acc0 = wmma_bf16(a, b0, acc0);
    acc1 = wmma_bf16(a, b1, acc1);
  }
  float bc0 = bias[c0], bc1 = bias[c1];
  int v = bv % VV, b = bv / VV;
  int n = b >> 1, mm = b & 1;
#pragma unroll
  for (int r = 0; r < 8; ++r) {
    int t = t0 + g * 8 + r;
    if (t < TT) {
      out[(((((size_t)n*CC + c0)*VV + v)*TT + t)*MM) + mm] = acc0[r] + bc0;
      out[(((((size_t)n*CC + c1)*VV + v)*TT + t)*MM) + mm] = acc1[r] + bc1;
    }
  }
}

// ---------------- launch ----------------
extern "C" void kernel_launch(void* const* d_in, const int* in_sizes, int n_in,
                              void* d_out, int out_size, void* d_ws, size_t ws_size,
                              hipStream_t stream) {
  const float* x       = (const float*)d_in[0];
  const float* w_qkv   = (const float*)d_in[1];
  const float* b_qkv   = (const float*)d_in[2];
  const float* w_merge = (const float*)d_in[3];
  const float* b_merge = (const float*)d_in[4];
  const float* bt      = (const float*)d_in[5];
  float* out = (float*)d_out;

  char* ws = (char*)d_ws;
  size_t off = 0;
  __bf16* xb  = (__bf16*)(ws + off); off += (size_t)BB*VV*TP*CC    * 2;  // 8.19 MB
  __bf16* qb  = (__bf16*)(ws + off); off += (size_t)BB*VV*HH*TP*HD * 2;  // 32.8 MB
  __bf16* kb  = (__bf16*)(ws + off); off += (size_t)BB*VV*HH*TP*HD * 2;  // 32.8 MB
  __bf16* vt  = (__bf16*)(ws + off); off += (size_t)BB*VV*HH*HD*TP * 2;  // 32.8 MB
  __bf16* ao  = (__bf16*)(ws + off); off += (size_t)BB*VV*TP*EE    * 2;  // 32.8 MB
  __bf16* wqt = (__bf16*)(ws + off); off += (size_t)768*64  * 2;
  __bf16* btb = (__bf16*)(ws + off); off += (size_t)BTP*HD  * 2;
  __bf16* wmt = (__bf16*)(ws + off); off += (size_t)CC*EE   * 2;
  (void)off; (void)ws_size; (void)in_sizes; (void)n_in; (void)out_size;

  // pack / transpose to bf16
  {
    int total = BB*VV*TP*CC;
    pack_x<<<(total + 255) / 256, 256, 0, stream>>>(x, xb);
    pack_wqkv<<<(768*64 + 255) / 256, 256, 0, stream>>>(w_qkv, wqt);
    pack_bt<<<(BTP*HD + 255) / 256, 256, 0, stream>>>(bt, btb);
    pack_wm<<<(CC*EE + 255) / 256, 256, 0, stream>>>(w_merge, wmt);
  }
  // stage 1: QKV projection (two col-tiles per wave)
  {
    int tiles = BB*VV*NT_T * 24;              // 91200 wave-tiles
    qkv_gemm<<<(tiles + 7) / 8, 256, 0, stream>>>(xb, wqt, b_qkv, qb, kb, vt);
  }
  // stage 2: attention
  {
    int blocks = BB*VV*HH * NT_T;             // 30400, one wave each
    attn_kernel<<<blocks, 32, 0, stream>>>(qb, kb, vt, btb, ao);
  }
  // stage 3: merge projection + output layout (two col-tiles per wave)
  {
    int tiles = BB*VV*NT_T * 2;               // 7600 wave-tiles
    merge_gemm<<<(tiles + 7) / 8, 256, 0, stream>>>(ao, wmt, b_merge, out);
  }
}
